// CausalAttention_88897233092997
// MI455X (gfx1250) — matrix operand order repaired and checked by reading, not compiled, not run on hardware
//
#include <hip/hip_runtime.h>

// ---------------------------------------------------------------------------
// CDNA5 (gfx1250) bf16 WMMA causal multi-head attention.
//  - wave32, V_WMMA_F32_16X16X32_BF16 for all matmuls
//  - Q/K stored [B,H,S,Dh]; V stored transposed [B,H,Dh,S] so every WMMA
//    B-fragment is a contiguous-K (2 x b128) load per lane.
//  - flash-attention with online softmax, P staged through per-wave LDS to
//    re-layout C-fragment scores into an A-fragment (s_wait_dscnt fence).
// ---------------------------------------------------------------------------

typedef __attribute__((ext_vector_type(16))) __bf16 v16bf;
typedef __attribute__((ext_vector_type(8)))  __bf16 bf16x8;
typedef __attribute__((ext_vector_type(8)))  float  v8f;

#define D_MODEL  1024
#define N_HEADS  16
#define HEAD_DIM 64
#define BATCH    2
#define SEQ      2048
#define MTOK     (BATCH * SEQ)   // 4096 tokens

__device__ __forceinline__ v8f wmma_bf16(v16bf a, v16bf b, v8f c) {
  // D = A(16x32) * B(32x16) + C, f32 accumulate
  return __builtin_amdgcn_wmma_f32_16x16x32_bf16(
      /*neg_a=*/false, a, /*neg_b=*/false, b,
      /*c_mod=*/(short)0, c, /*reuse_a=*/false, /*reuse_b=*/false);
}

// A fragment (16x32 bf16, MxK): lane m (0..15) of half h holds rows m,
// K runs [h*8, h*8+8) and [16+h*8, 16+h*8+8): two contiguous 16B chunks.
template <typename PT>
__device__ __forceinline__ v16bf load_a_frag(PT base, int ld, int mBase,
                                             int kBase, int lane) {
  const int h = lane >> 4, m = lane & 15;
  union { v16bf v; bf16x8 p[2]; } u;
  PT r = base + (size_t)(mBase + m) * ld + kBase + h * 8;
  u.p[0] = *reinterpret_cast<const bf16x8*>(r);
  u.p[1] = *reinterpret_cast<const bf16x8*>(r + 16);
  return u.v;
}

// B fragment (32x16 bf16, KxN) from a K-major matrix M[n][k] (ld = K stride):
// lane n of half h holds K runs [h*8, h*8+8) and [16+h*8, 16+h*8+8)
// (same order as the A fragment): two 16B chunks.
__device__ __forceinline__ v16bf load_b_frag(const __bf16* base, int ld,
                                             int nBase, int kBase, int lane) {
  const int h = lane >> 4, n = lane & 15;
  union { v16bf v; bf16x8 p[2]; } u;
  const __bf16* r = base + (size_t)(nBase + n) * ld + kBase + h * 8;
  u.p[0] = *reinterpret_cast<const bf16x8*>(r);
  u.p[1] = *reinterpret_cast<const bf16x8*>(r + 16);
  return u.v;
}

// ---------------------------------------------------------------------------
// fp32 -> bf16 conversion of activations
__global__ __launch_bounds__(256) void cvt_f32_bf16(const float* __restrict__ x,
                                                    __bf16* __restrict__ y,
                                                    int n) {
  int i = blockIdx.x * blockDim.x + threadIdx.x;
  if (i < n) y[i] = (__bf16)x[i];
}

// fp32 -> bf16 transposed convert: Wt[n][k] = W[k][n], via 32x32 LDS tile.
__global__ __launch_bounds__(256) void cvt_w_transpose(const float* __restrict__ W,
                                                       __bf16* __restrict__ Wt) {
  __shared__ __align__(16) __bf16 tile[32][33];
  const int bx = blockIdx.x * 32, by = blockIdx.y * 32;
  const int tx = threadIdx.x, ty = threadIdx.y;  // block (32, 8)
  for (int i = 0; i < 32; i += 8)
    tile[ty + i][tx] = (__bf16)W[(size_t)(by + ty + i) * D_MODEL + bx + tx];
  __syncthreads();
  for (int i = 0; i < 32; i += 8)
    Wt[(size_t)(bx + ty + i) * D_MODEL + by + tx] = tile[tx][ty + i];
}

// ---------------------------------------------------------------------------
// bf16 GEMM: Y[M=4096, N=1024] = X[M,K=1024] * Wt[N,K]^T + bias.
// One wave computes a 32x64 tile (2x4 WMMA tiles), K-loop step 32, 8 WMMA/iter.
// mode 0: Y -> Q buffer [B,H,S,Dh] (bf16)
// mode 1: Y -> K buffer [B,H,S,Dh] (bf16)
// mode 2: Y -> V^T buffer [B,H,Dh,S] (bf16)
// mode 3: Y -> fp32 row-major [M, N] (final output)
__global__ __launch_bounds__(256) void gemm_bf16(const __bf16* __restrict__ X,
                                                 const __bf16* __restrict__ Wt,
                                                 const float* __restrict__ bias,
                                                 int mode,
                                                 void* __restrict__ outp) {
  const int lane = threadIdx.x & 31;
  const int wid  = (blockIdx.x << 3) + (threadIdx.x >> 5);
  const int nTiles = D_MODEL / 64;           // 16 column blocks
  const int mBase = (wid / nTiles) * 32;
  const int nBase = (wid % nTiles) * 64;

  v8f acc[2][4] = {};
  for (int k0 = 0; k0 < D_MODEL; k0 += 32) {
    v16bf a0 = load_a_frag(X, D_MODEL, mBase,      k0, lane);
    v16bf a1 = load_a_frag(X, D_MODEL, mBase + 16, k0, lane);
    v16bf b0 = load_b_frag(Wt, D_MODEL, nBase,      k0, lane);
    v16bf b1 = load_b_frag(Wt, D_MODEL, nBase + 16, k0, lane);
    v16bf b2 = load_b_frag(Wt, D_MODEL, nBase + 32, k0, lane);
    v16bf b3 = load_b_frag(Wt, D_MODEL, nBase + 48, k0, lane);
    acc[0][0] = wmma_bf16(a0, b0, acc[0][0]);
    acc[0][1] = wmma_bf16(a0, b1, acc[0][1]);
    acc[0][2] = wmma_bf16(a0, b2, acc[0][2]);
    acc[0][3] = wmma_bf16(a0, b3, acc[0][3]);
    acc[1][0] = wmma_bf16(a1, b0, acc[1][0]);
    acc[1][1] = wmma_bf16(a1, b1, acc[1][1]);
    acc[1][2] = wmma_bf16(a1, b2, acc[1][2]);
    acc[1][3] = wmma_bf16(a1, b3, acc[1][3]);
  }

  const int h = lane >> 4, nl = lane & 15;
  for (int i = 0; i < 2; ++i)
    for (int j = 0; j < 4; ++j)
      for (int e = 0; e < 8; ++e) {
        const int row = mBase + i * 16 + e + 8 * h;   // token index
        const int col = nBase + j * 16 + nl;          // feature index
        float v = acc[i][j][e] + bias[col];
        const int b = row >> 11, s = row & (SEQ - 1);
        const int hd = col >> 6, d = col & 63;
        if (mode == 3) {
          ((float*)outp)[(size_t)row * D_MODEL + col] = v;
        } else if (mode == 2) {
          ((__bf16*)outp)[((size_t)(b * N_HEADS + hd) * HEAD_DIM + d) * SEQ + s] =
              (__bf16)v;
        } else {
          ((__bf16*)outp)[((size_t)(b * N_HEADS + hd) * SEQ + s) * HEAD_DIM + d] =
              (__bf16)v;
        }
      }
}

// ---------------------------------------------------------------------------
// Flash attention: one wave owns (b,h, 16-query tile); streams 32-key blocks.
// Per block: 4 WMMA for Q*K^T, online softmax, P staged via LDS, 4 WMMA P*V.
__global__ __launch_bounds__(256) void attn_flash(const __bf16* __restrict__ Q,
                                                  const __bf16* __restrict__ K,
                                                  const __bf16* __restrict__ Vt,
                                                  __bf16* __restrict__ Hout) {
  __shared__ __align__(16) __bf16 pTile[8][16 * 32];   // per-wave P scratch
  const int lane = threadIdx.x & 31;
  const int wave = threadIdx.x >> 5;
  const int wid  = (blockIdx.x << 3) + wave;
  const int qTile = wid & (SEQ / 16 - 1);   // 0..127
  const int bh    = wid >> 7;               // 0..31 = b*16+head
  const int qBase = qTile * 16;
  const int h = lane >> 4, nl = lane & 15;

  const __bf16* Qp = Q  + (size_t)bh * SEQ * HEAD_DIM;
  const __bf16* Kp = K  + (size_t)bh * SEQ * HEAD_DIM;
  const __bf16* Vp = Vt + (size_t)bh * HEAD_DIM * SEQ;

  // Q fragments for the whole tile (Dh = 64 -> two k-steps)
  const v16bf aq0 = load_a_frag(Qp, HEAD_DIM, qBase, 0,  lane);
  const v16bf aq1 = load_a_frag(Qp, HEAD_DIM, qBase, 32, lane);

  v8f acc[4] = {};                    // 16 queries x 64 dims
  float mrow[8], lrow[8];
#pragma unroll
  for (int e = 0; e < 8; ++e) { mrow[e] = -1e30f; lrow[e] = 0.0f; }
  const float scale = 0.125f;         // 1/sqrt(64)

  const int nKeys = qBase + 16;       // causal horizon for this tile
  for (int kb = 0; kb < nKeys; kb += 32) {
    // ---- scores: two 16-key sub-tiles, each over Dh=64 (2 chained WMMAs)
    v8f s0 = {}, s1 = {};
    {
      v16bf b0 = load_b_frag(Kp, HEAD_DIM, kb, 0,  lane);
      v16bf b1 = load_b_frag(Kp, HEAD_DIM, kb, 32, lane);
      s0 = wmma_bf16(aq0, b0, s0);
      s0 = wmma_bf16(aq1, b1, s0);
    }
    {
      v16bf b0 = load_b_frag(Kp, HEAD_DIM, kb + 16, 0,  lane);
      v16bf b1 = load_b_frag(Kp, HEAD_DIM, kb + 16, 32, lane);
      s1 = wmma_bf16(aq0, b0, s1);
      s1 = wmma_bf16(aq1, b1, s1);
    }

    // ---- online softmax update (rows = e + 8*h, col = key lane)
    const int k0 = kb + nl, k1 = kb + 16 + nl;
    float p0[8], p1[8];
#pragma unroll
    for (int e = 0; e < 8; ++e) {
      const int q = qBase + e + 8 * h;
      float v0 = s0[e] * scale; if (k0 > q) v0 = -1e30f;
      float v1 = s1[e] * scale; if (k1 > q) v1 = -1e30f;
      float mx = fmaxf(v0, v1);
#pragma unroll
      for (int off = 1; off < 16; off <<= 1)
        mx = fmaxf(mx, __shfl_xor(mx, off, 32));
      const float mnew = fmaxf(mrow[e], mx);
      const float corr = __expf(mrow[e] - mnew);
      v0 = __expf(v0 - mnew);
      v1 = __expf(v1 - mnew);
      float rs = v0 + v1;
#pragma unroll
      for (int off = 1; off < 16; off <<= 1)
        rs += __shfl_xor(rs, off, 32);
      lrow[e] = lrow[e] * corr + rs;
      mrow[e] = mnew;
#pragma unroll
      for (int j = 0; j < 4; ++j) acc[j][e] *= corr;
      p0[e] = v0;
      p1[e] = v1;
    }

    // ---- re-layout P (C-fragment) into A-fragment via per-wave LDS
    __bf16* pt = &pTile[wave][0];
#pragma unroll
    for (int e = 0; e < 8; ++e) {
      const int r = e + 8 * h;
      pt[r * 32 + nl]      = (__bf16)p0[e];
      pt[r * 32 + 16 + nl] = (__bf16)p1[e];
    }
    asm volatile("s_wait_dscnt 0" ::: "memory");  // wave-synchronous LDS RAW
    const v16bf pa = load_a_frag((const __bf16*)pt, 32, 0, 0, lane);

    // ---- P(16x32) * V(32x64): 4 WMMAs; V^T gives contiguous-K B fragments
#pragma unroll
    for (int j = 0; j < 4; ++j) {
      v16bf bv = load_b_frag(Vp, SEQ, j * 16, kb, lane);
      acc[j] = wmma_bf16(pa, bv, acc[j]);
    }
  }

  // ---- normalize and store bf16 H tile: [token][head*64 + dim]
  const int b = bh >> 4, hd = bh & 15;
  const size_t row0 = (size_t)b * SEQ + qBase;
#pragma unroll
  for (int j = 0; j < 4; ++j)
#pragma unroll
    for (int e = 0; e < 8; ++e) {
      const int r = e + 8 * h;
      Hout[(row0 + r) * D_MODEL + hd * 64 + j * 16 + nl] =
          (__bf16)(acc[j][e] / lrow[e]);
    }
}

// ---------------------------------------------------------------------------
extern "C" void kernel_launch(void* const* d_in, const int* in_sizes, int n_in,
                              void* d_out, int out_size, void* d_ws, size_t ws_size,
                              hipStream_t stream) {
  (void)in_sizes; (void)n_in; (void)out_size; (void)ws_size;
  const float* x  = (const float*)d_in[0];
  const float* Wq = (const float*)d_in[1];
  const float* bq = (const float*)d_in[2];
  const float* Wk = (const float*)d_in[3];
  const float* bk = (const float*)d_in[4];
  const float* Wv = (const float*)d_in[5];
  const float* bv = (const float*)d_in[6];
  const float* Wo = (const float*)d_in[7];
  const float* bo = (const float*)d_in[8];

  char* ws = (char*)d_ws;
  const size_t SZ_X   = (size_t)MTOK * D_MODEL * sizeof(__bf16);            // 8 MiB
  const size_t SZ_W   = (size_t)D_MODEL * D_MODEL * sizeof(__bf16);         // 2 MiB
  const size_t SZ_QKV = (size_t)BATCH * N_HEADS * SEQ * HEAD_DIM * sizeof(__bf16);

  __bf16* xb  = (__bf16*)(ws);                 size_t off = SZ_X;
  __bf16* Wqt = (__bf16*)(ws + off); off += SZ_W;
  __bf16* Wkt = (__bf16*)(ws + off); off += SZ_W;
  __bf16* Wvt = (__bf16*)(ws + off); off += SZ_W;
  __bf16* Wot = (__bf16*)(ws + off); off += SZ_W;
  __bf16* Qb  = (__bf16*)(ws + off); off += SZ_QKV;
  __bf16* Kb  = (__bf16*)(ws + off); off += SZ_QKV;
  __bf16* Vtb = (__bf16*)(ws + off); off += SZ_QKV;
  __bf16* Hb  = (__bf16*)(ws + off); off += SZ_X;

  // 1) convert activations + transposed weights to bf16
  {
    const int n = MTOK * D_MODEL;
    cvt_f32_bf16<<<(n + 255) / 256, 256, 0, stream>>>(x, xb, n);
    dim3 g(D_MODEL / 32, D_MODEL / 32), blk(32, 8);
    cvt_w_transpose<<<g, blk, 0, stream>>>(Wq, Wqt);
    cvt_w_transpose<<<g, blk, 0, stream>>>(Wk, Wkt);
    cvt_w_transpose<<<g, blk, 0, stream>>>(Wv, Wvt);
    cvt_w_transpose<<<g, blk, 0, stream>>>(Wo, Wot);
  }

  // 2) QKV projections (WMMA GEMMs)
  const int gemmBlocks = (MTOK / 32) * (D_MODEL / 64) / 8;   // 256
  gemm_bf16<<<gemmBlocks, 256, 0, stream>>>(xb, Wqt, bq, 0, Qb);
  gemm_bf16<<<gemmBlocks, 256, 0, stream>>>(xb, Wkt, bk, 1, Kb);
  gemm_bf16<<<gemmBlocks, 256, 0, stream>>>(xb, Wvt, bv, 2, Vtb);

  // 3) causal flash attention (WMMA)
  const int attnBlocks = (BATCH * N_HEADS) * (SEQ / 16) / 8;  // 512
  attn_flash<<<attnBlocks, 256, 0, stream>>>(Qb, Kb, Vtb, Hb);

  // 4) output projection, fp32 + bias -> d_out
  gemm_bf16<<<gemmBlocks, 256, 0, stream>>>(Hb, Wot, bo, 3, (float*)d_out);
}
